// RSA_MACE_15573551415371
// MI455X (gfx1250) — matrix operand order repaired and checked by reading, not compiled, not run on hardware
//
#include <hip/hip_runtime.h>
#include <hip/hip_bf16.h>
#include <math.h>

typedef __attribute__((ext_vector_type(16))) __bf16 v16bf;
typedef __attribute__((ext_vector_type(8)))  float  v8f;

#define NN 4096
#define HH 64
#define TT 6
#define GG 16

// ---------------------------------------------------------------------------
// K0: fold the two linears: W_comb[192][64] = W_qkv(192x64) @ W_readout(64x64)
// ---------------------------------------------------------------------------
__global__ void kA_wcomb(const float* __restrict__ wqkv,
                         const float* __restrict__ wread,
                         float* __restrict__ wcomb) {
    int id = blockIdx.x * 256 + threadIdx.x;           // < 192*64
    int m = id >> 6, c = id & 63;
    float s = 0.f;
    #pragma unroll 4
    for (int h = 0; h < 64; ++h) s += wqkv[m * 64 + h] * wread[h * 64 + c];
    wcomb[id] = s;
}

// ---------------------------------------------------------------------------
// K1: segment offsets from sorted batch (binary search, 17 entries)
// ---------------------------------------------------------------------------
__global__ void kOffs(const int* __restrict__ batch, int* __restrict__ offs) {
    int t = threadIdx.x;
    if (t > 16) return;
    int lo = 0, hi = NN;
    while (lo < hi) { int mid = (lo + hi) >> 1; if (batch[mid] < t) lo = mid + 1; else hi = mid; }
    offs[t] = lo;
}

// ---------------------------------------------------------------------------
// K2: pre-swizzle W_tp into bf16 B-fragments for v_wmma_f32_16x16x32_bf16.
// B layout used here (32x16 KxN, k order matches A): lanes 0-15 hold N=lane, K=t (t=0..15); lanes 16-31
// hold N=lane-16, K=16+t. Stored so each lane reads 16 contiguous bf16.
// Flat: bswz[((c*4 + w)*32 + l)*16 + t]  with kdim = 32c + (l>>4)*16 + t,
//       i = kdim>>6, j = kdim&63, kout = 16w + (l&15);  value = W_tp[kout][i][j]
// ---------------------------------------------------------------------------
__global__ void kW_prep_b(const float* __restrict__ wtp, __bf16* __restrict__ bswz) {
    int id = blockIdx.x * 256 + threadIdx.x;           // < 262144
    int t = id & 15;
    int l = (id >> 4) & 31;
    int w = (id >> 9) & 3;
    int c = id >> 11;
    int kdim = c * 32 + ((l >> 4) << 4) + t;
    int i = kdim >> 6, j = kdim & 63;
    int kout = w * 16 + (l & 15);
    bswz[id] = (__bf16)wtp[(kout * 64 + i) * 64 + j];
}

// ---------------------------------------------------------------------------
// K3: qkv (via W_comb) + SiLU + RoPE.  One block per node, 64 threads.
// q_rope/k_rope stored as [t][n][h] fp32.
// ---------------------------------------------------------------------------
__global__ __launch_bounds__(64) void kB_qkv_rope(
    const float* __restrict__ nf, const float* __restrict__ pos,
    const float* __restrict__ kvecs, const int* __restrict__ batch,
    const float* __restrict__ wcomb,
    float* __restrict__ qrope, float* __restrict__ krope, float* __restrict__ vws) {
    __shared__ float nfS[64], qS[64], kS[64];
    const int n = blockIdx.x, h = threadIdx.x;
    nfS[h] = nf[n * 64 + h];
    __syncthreads();
    float q = 0.f, k = 0.f, v = 0.f;
    #pragma unroll 4
    for (int c2 = 0; c2 < 64; ++c2) {
        float x = nfS[c2];
        q += x * wcomb[h * 64 + c2];
        k += x * wcomb[(64 + h) * 64 + c2];
        v += x * wcomb[(128 + h) * 64 + c2];
    }
    qS[h] = q / (1.f + __expf(-q));                    // silu
    kS[h] = k / (1.f + __expf(-k));
    vws[n * 64 + h] = v;                               // v: no silu
    __syncthreads();
    const int g = batch[n];
    const float px = pos[n * 3], py = pos[n * 3 + 1], pz = pos[n * 3 + 2];
    const int ia = (h < 32) ? 2 * h : 2 * (h - 32);
    const float aq = qS[ia], bq = qS[ia + 1];
    const float ak = kS[ia], bk = kS[ia + 1];
    for (int t = 0; t < TT; ++t) {
        const float* kvt = &kvecs[(g * TT + t) * 3];
        float ph = px * kvt[0] + py * kvt[1] + pz * kvt[2];
        float sn, cs;
        __sincosf(ph, &sn, &cs);
        float qr, kr;
        if (h < 32) { qr = aq * cs - bq * sn; kr = ak * cs - bk * sn; }
        else        { qr = aq * sn + bq * cs; kr = ak * sn + bk * cs; }
        qrope[((t * NN) + n) * 64 + h] = qr * 0.125f;  // scale_q = 1/sqrt(64)
        krope[((t * NN) + n) * 64 + h] = kr;
    }
}

// ---------------------------------------------------------------------------
// K4: kv_graph[g][t][d][e] = sum_{n in g} k[t,n,d]*v[n,e].  Block per (g,t),
// 256 threads, each owns 16 (d,e) accumulators; nodes staged in LDS.
// ---------------------------------------------------------------------------
__global__ __launch_bounds__(256) void kC_kvgraph(
    const float* __restrict__ krope, const float* __restrict__ vws,
    const int* __restrict__ offs, float* __restrict__ kvg) {
    __shared__ float kS[64 * 64];
    __shared__ float vS[64 * 64];
    const int g = blockIdx.x / TT, t = blockIdx.x % TT;
    const int tid = threadIdx.x;
    const int e = tid & 63, dbase = (tid >> 6) << 4;
    const int start = offs[g], end = offs[g + 1];
    float acc[16];
    #pragma unroll
    for (int i = 0; i < 16; ++i) acc[i] = 0.f;
    for (int base = start; base < end; base += 64) {
        const int count = min(64, end - base);
        for (int idx = tid; idx < 4096; idx += 256) {
            int nn = idx >> 6, d = idx & 63;
            int n = base + nn;
            kS[idx] = (nn < count) ? krope[((t * NN) + n) * 64 + d] : 0.f;
            vS[idx] = (nn < count) ? vws[n * 64 + d] : 0.f;
        }
        __syncthreads();
        for (int nn = 0; nn < count; ++nn) {
            float vv = vS[nn * 64 + e];
            #pragma unroll
            for (int i = 0; i < 16; ++i) acc[i] += kS[nn * 64 + dbase + i] * vv;
        }
        __syncthreads();
    }
    #pragma unroll
    for (int i = 0; i < 16; ++i)
        kvg[((g * TT + t) * 64 + dbase + i) * 64 + e] = acc[i];
}

// ---------------------------------------------------------------------------
// K5: update[n][e] = sum_{t,d} q[t,n,d]*kvg[batch[n]][t][d][e].
// 32 nodes per block, q staged in LDS; kvg value reused across the 8 nodes
// a thread owns (reloaded only on graph boundary).
// ---------------------------------------------------------------------------
__global__ __launch_bounds__(256) void kD_update(
    const float* __restrict__ qrope, const float* __restrict__ kvg,
    const int* __restrict__ batch, float* __restrict__ upd) {
    __shared__ float qS[32 * 384];
    __shared__ int gS[32];
    const int n0 = blockIdx.x * 32;
    const int tid = threadIdx.x;
    if (tid < 32) gS[tid] = batch[n0 + tid];
    for (int idx = tid; idx < 32 * 384; idx += 256) {
        int nn = idx / 384, r = idx - nn * 384;
        int t = r >> 6, h = r & 63;
        qS[idx] = qrope[((t * NN) + n0 + nn) * 64 + h];
    }
    __syncthreads();
    const int e = tid & 63, nslot = tid >> 6;          // 4 node groups of 8
    float acc[8];
    #pragma unroll
    for (int i = 0; i < 8; ++i) acc[i] = 0.f;
    for (int td = 0; td < 384; ++td) {
        int curg = -1; float kvv = 0.f;
        #pragma unroll
        for (int i = 0; i < 8; ++i) {
            int nn = nslot * 8 + i;
            int g = gS[nn];
            if (g != curg) { curg = g; kvv = kvg[(g * 384 + td) * 64 + e]; }
            acc[i] += qS[nn * 384 + td] * kvv;
        }
    }
    #pragma unroll
    for (int i = 0; i < 8; ++i)
        upd[(n0 + nslot * 8 + i) * 64 + e] = acc[i];
}

// ---------------------------------------------------------------------------
// K6 (dominant, WMMA): out(4096x64) = A(4096x4096) @ B(4096x64)
//   A[n, i*64+j] = sr[n,i]*update[n,j]   -- generated in registers per chunk
//   B pre-swizzled bf16 (kW_prep_b).  Block = 64 nodes, 4 waves; each wave:
//   16 nodes x 64 outputs = 4 accumulators, 128 K-chunks x 4 WMMAs = 512 WMMAs.
// A-frag 16x32 bf16 layout used here (k order matches B): lane (khalf*16+row) holds M=row,
//   element t -> K = t + khalf*16.
// ---------------------------------------------------------------------------
__global__ __launch_bounds__(128) void kE_tp_wmma(
    const float* __restrict__ sr, const float* __restrict__ upd,
    const __bf16* __restrict__ bswz, float* __restrict__ out) {
    __shared__ float srS[64 * 65];                     // stride 65: no bank conflicts
    __shared__ float updS[64 * 65];
    const int n0 = blockIdx.x * 64;
    const int tid = threadIdx.x;
    for (int idx = tid; idx < 4096; idx += 128) {
        int r = idx >> 6, c = idx & 63;
        srS[r * 65 + c]  = sr[n0 * 64 + idx];
        updS[r * 65 + c] = upd[n0 * 64 + idx];
    }
    __syncthreads();

    const int lane  = tid & 31;
    const int wv    = tid >> 5;
    const int row   = lane & 15;
    const int khalf = lane >> 4;
    const int nl    = wv * 16 + row;

    v8f acc[4];
    #pragma unroll
    for (int i = 0; i < 4; ++i) acc[i] = (v8f){0.f, 0.f, 0.f, 0.f, 0.f, 0.f, 0.f, 0.f};

    for (int c = 0; c < 128; ++c) {
        const int i  = c >> 1;
        const int j0 = (c & 1) << 5;
        const float srv = srS[nl * 65 + i];
        const float* ur = &updS[nl * 65 + j0];
        v16bf a;
        #pragma unroll
        for (int t = 0; t < 16; ++t) {
            const int K = t + (khalf << 4);
            a[t] = (__bf16)(srv * ur[K]);
        }
        const __bf16* bp = bswz + ((size_t)c * 128 + lane) * 16;
        if (c + 2 < 128)
            __builtin_prefetch(bswz + ((size_t)(c + 2) * 128 + lane) * 16, 0, 0);
        #pragma unroll
        for (int nt = 0; nt < 4; ++nt) {
            v16bf b = *(const v16bf*)(bp + nt * 512);  // 32 lanes * 16 elems
            acc[nt] = __builtin_amdgcn_wmma_f32_16x16x32_bf16(
                false, a, false, b, (short)0, acc[nt], false, false);
        }
    }

    #pragma unroll
    for (int nt = 0; nt < 4; ++nt) {
        #pragma unroll
        for (int r = 0; r < 8; ++r) {
            const int m = khalf * 8 + r;               // C/D layout: M = r + 8*khalf
            out[(size_t)(n0 + wv * 16 + m) * 64 + nt * 16 + row] = acc[nt][r];
        }
    }
}

// ---------------------------------------------------------------------------
// Launch. Workspace layout (bytes, all 256-aligned):
//   wcomb 0..49152 | offs 49152 | qrope 49408 (+6291456) | krope 6340864
//   | v 12632320 (+1048576) | kvg 13680896 (+1572864) | upd 15253760 (+1048576)
//   | bswz 16302336 (+524288)   total ~16.8 MB
// ---------------------------------------------------------------------------
extern "C" void kernel_launch(void* const* d_in, const int* in_sizes, int n_in,
                              void* d_out, int out_size, void* d_ws, size_t ws_size,
                              hipStream_t stream) {
    const float* node_feat    = (const float*)d_in[0];
    const float* node_feat_sr = (const float*)d_in[1];
    const float* positions    = (const float*)d_in[2];
    const float* kvecs        = (const float*)d_in[3];
    const int*   batch        = (const int*)d_in[4];
    const float* W_readout    = (const float*)d_in[5];
    const float* W_qkv        = (const float*)d_in[6];
    const float* W_tp         = (const float*)d_in[7];
    float* out = (float*)d_out;

    char* ws = (char*)d_ws;
    float*  wcomb = (float*)(ws + 0);
    int*    offs  = (int*)  (ws + 49152);
    float*  qrope = (float*)(ws + 49408);
    float*  krope = (float*)(ws + 6340864);
    float*  vws   = (float*)(ws + 12632320);
    float*  kvg   = (float*)(ws + 13680896);
    float*  upd   = (float*)(ws + 15253760);
    __bf16* bswz  = (__bf16*)(ws + 16302336);

    kA_wcomb   <<<48,   256, 0, stream>>>(W_qkv, W_readout, wcomb);
    kOffs      <<<1,    32,  0, stream>>>(batch, offs);
    kW_prep_b  <<<1024, 256, 0, stream>>>(W_tp, bswz);
    kB_qkv_rope<<<NN,   64,  0, stream>>>(node_feat, positions, kvecs, batch,
                                          wcomb, qrope, krope, vws);
    kC_kvgraph <<<GG * TT, 256, 0, stream>>>(krope, vws, offs, kvg);
    kD_update  <<<NN / 32, 256, 0, stream>>>(qrope, kvg, batch, upd);
    kE_tp_wmma <<<NN / 64, 128, 0, stream>>>(node_feat_sr, upd, bswz, out);
}
